// TriangleMultiplication_21440476741691
// MI455X (gfx1250) — matrix operand order repaired and checked by reading, not compiled, not run on hardware
//
#include <hip/hip_runtime.h>
#include <hip/hip_bf16.h>

#define NDIM 512
#define CDIM 128
#define NNTOT (NDIM * NDIM)   // 262144 rows

typedef __attribute__((ext_vector_type(8)))  __bf16 v8bf;
typedef __attribute__((ext_vector_type(16))) __bf16 v16bf;
typedef __attribute__((ext_vector_type(8)))  float  v8f;

union BF16Frag { v16bf v; v8bf h[2]; };

__device__ __forceinline__ unsigned short f2bf_raw(float f) {
  union { float f; unsigned u; } v; v.f = f;
  unsigned r = v.u + 0x7FFFu + ((v.u >> 16) & 1u);   // round-to-nearest-even
  return (unsigned short)(r >> 16);
}
__device__ __forceinline__ float bf2f_raw(unsigned short h) {
  union { unsigned u; float f; } v; v.u = ((unsigned)h) << 16;
  return v.f;
}
__device__ __forceinline__ float sigmoidf_(float x) {
  return 1.0f / (1.0f + __expf(-x));
}

// A-operand fragment (16x32 bf16, M in lanes 0..15 mirrored in 16..31,
// K pairs: lane-half 0 -> K {k0..k0+7, k0+16..k0+23}, half 1 -> +8).
__device__ __forceinline__ v16bf load_a_frag(const unsigned short* base, int ld,
                                             int row0, int k0, int lane) {
  int r = lane & 15, half = lane >> 4;
  const unsigned short* p = base + (size_t)(row0 + r) * ld + k0 + half * 8;
  BF16Frag f;
  f.h[0] = *(const v8bf*)p;
  f.h[1] = *(const v8bf*)(p + 16);
  return f.v;
}

// B-operand fragment (32x16 bf16, N in lanes; lane-half 0 -> K {0..7,16..23}, half 1 -> +8).
// Memory is [n][k] row-major (so the GEMM computes A * B^T): 2x16B per lane.
__device__ __forceinline__ v16bf load_b_frag(const unsigned short* base, int ld,
                                             int col0, int k0, int lane) {
  int r = lane & 15, half = lane >> 4;
  const unsigned short* p = base + (size_t)(col0 + r) * ld + k0 + half * 8;
  BF16Frag f;
  f.h[0] = *(const v8bf*)p;
  f.h[1] = *(const v8bf*)(p + 16);
  return f.v;
}

__device__ __forceinline__ v8f wmma_bf16(v16bf a, v16bf b, v8f c) {
  return __builtin_amdgcn_wmma_f32_16x16x32_bf16(false, a, false, b, (short)0, c,
                                                 false, false);
}

// ---------------------------------------------------------------- kernel 0
// Convert weights to bf16: [Wg(256x128) | Wp(256x128) | Wo(128x128) | Wpout(128x128)]
__global__ void __launch_bounds__(256) k_prep(const float* __restrict__ g_in_w,
                                              const float* __restrict__ p_in_w,
                                              const float* __restrict__ g_out_w,
                                              const float* __restrict__ p_out_w,
                                              unsigned short* __restrict__ wbf) {
  int idx = blockIdx.x * 256 + threadIdx.x;   // 0..98303
  float v;
  if (idx < 32768)       v = g_in_w[idx];
  else if (idx < 65536)  v = p_in_w[idx - 32768];
  else if (idx < 81920)  v = g_out_w[idx - 65536];
  else                   v = p_out_w[idx - 81920];
  wbf[idx] = f2bf_raw(v);
}

// ---------------------------------------------------------------- kernel 1
// LayerNorm(x) -> xn bf16, one wave32 per 128-channel row.
__global__ void __launch_bounds__(128) k_ln_in(const float* __restrict__ x,
                                               const float* __restrict__ w,
                                               const float* __restrict__ b,
                                               unsigned short* __restrict__ xn) {
  int wave = threadIdx.x >> 5, lane = threadIdx.x & 31;
  size_t row = (size_t)blockIdx.x * 4 + wave;
  float4 xv = ((const float4*)(x + row * CDIM))[lane];
  float s  = xv.x + xv.y + xv.z + xv.w;
  float s2 = xv.x * xv.x + xv.y * xv.y + xv.z * xv.z + xv.w * xv.w;
#pragma unroll
  for (int m = 16; m >= 1; m >>= 1) {
    s  += __shfl_xor(s,  m, 32);
    s2 += __shfl_xor(s2, m, 32);
  }
  float mu   = s  * (1.0f / CDIM);
  float var  = s2 * (1.0f / CDIM) - mu * mu;
  float rstd = rsqrtf(var + 1e-5f);
  float4 wv = ((const float4*)w)[lane];
  float4 bv = ((const float4*)b)[lane];
  ushort4 o;
  o.x = f2bf_raw((xv.x - mu) * rstd * wv.x + bv.x);
  o.y = f2bf_raw((xv.y - mu) * rstd * wv.y + bv.y);
  o.z = f2bf_raw((xv.z - mu) * rstd * wv.z + bv.z);
  o.w = f2bf_raw((xv.w - mu) * rstd * wv.w + bv.w);
  ((ushort4*)(xn + row * CDIM))[lane] = o;
}

// ---------------------------------------------------------------- kernel 2
// Fused projections, A-stationary: each wave loads its 16x128 A slice ONCE
// (4 fragments in registers), then sweeps all 16 n-tiles of
// G = xn*Wg^T, P = xn*Wp^T, O = xn*Wo^T (weights are L2-resident, 192KB).
// Epilogue: pg = P*sigmoid(G) scattered to a_t/b_t ([c][m] bf16, packed 16B
// stores: 8 consecutive m per lane), gate = sigmoid(O) stored [m][c] bf16.
__global__ void __launch_bounds__(128) k_proj(const unsigned short* __restrict__ xn,
                                              const unsigned short* __restrict__ wbf,
                                              unsigned short* __restrict__ at,
                                              unsigned short* __restrict__ bt,
                                              unsigned short* __restrict__ gate) {
  const unsigned short* Wg = wbf;
  const unsigned short* Wp = wbf + 32768;
  const unsigned short* Wo = wbf + 65536;
  int wave = threadIdx.x >> 5, lane = threadIdx.x & 31;
  int m0 = (blockIdx.x * 4 + wave) * 16;
  int half = lane >> 4, col = lane & 15;

  // A fragments for K = 0..127, loaded once and reused for all 16 n-tiles.
  v16bf a0 = load_a_frag(xn, CDIM, m0, 0,  lane);
  v16bf a1 = load_a_frag(xn, CDIM, m0, 32, lane);
  v16bf a2 = load_a_frag(xn, CDIM, m0, 64, lane);
  v16bf a3 = load_a_frag(xn, CDIM, m0, 96, lane);

  for (int nt = 0; nt < 16; ++nt) {
    int n0 = nt * 16;
    bool doO = (n0 < CDIM);           // uniform across block
    v8f accG = {}, accP = {}, accO = {};
#pragma unroll
    for (int kk = 0; kk < 4; ++kk) {
      v16bf a = (kk == 0) ? a0 : (kk == 1) ? a1 : (kk == 2) ? a2 : a3;
      int kt = kk * 32;
      accG = wmma_bf16(a, load_b_frag(Wg, CDIM, n0, kt, lane), accG);
      accP = wmma_bf16(a, load_b_frag(Wp, CDIM, n0, kt, lane), accP);
      if (doO)
        accO = wmma_bf16(a, load_b_frag(Wo, CDIM, n0, kt, lane), accO);
    }
    int cg = n0 + col;                // global output dim 0..255
    union { unsigned short s[8]; uint4 q; } pk;
#pragma unroll
    for (int r = 0; r < 8; ++r)
      pk.s[r] = f2bf_raw(accP[r] * sigmoidf_(accG[r]));
    // 8 consecutive m per lane -> one 16B store into the [c][m] transpose.
    unsigned short* dst = (cg < CDIM ? at + (size_t)cg * NNTOT
                                     : bt + (size_t)(cg - CDIM) * NNTOT)
                          + m0 + half * 8;
    *(uint4*)dst = pk.q;
    if (doO) {
#pragma unroll
      for (int r = 0; r < 8; ++r)
        gate[(size_t)(m0 + r + half * 8) * CDIM + cg] =
            f2bf_raw(sigmoidf_(accO[r]));
    }
  }
}

// ---------------------------------------------------------------- kernel 3
// tri[i,j,c] = sum_k a[i,k,c]*b[j,k,c]: per-channel 512x512x512 GEMM A_c*B_c^T.
// Wave register tile = 32x64 (2 A frags x 4 B frags -> 8 accumulators, 8 WMMA
// per 6 fragment loads). Block = 4 waves = 128x64 tile of one channel.
// Output channel-major [c][i][j] so lanes store consecutive j (coalesced).
__global__ void __launch_bounds__(128) k_tri(const unsigned short* __restrict__ at,
                                             const unsigned short* __restrict__ bt,
                                             float* __restrict__ triT) {
  int wave = threadIdx.x >> 5, lane = threadIdx.x & 31;
  int c  = blockIdx.y;
  int i0 = (blockIdx.x >> 3) * 128 + wave * 32;
  int jt = (blockIdx.x & 7) * 64;
  const unsigned short* A = at + (size_t)c * NNTOT;   // [i][k], ld=512
  const unsigned short* B = bt + (size_t)c * NNTOT;   // [j][k], ld=512
  v8f acc[2][4] = {};
  for (int kt = 0; kt < NDIM; kt += 32) {
    v16bf a0 = load_a_frag(A, NDIM, i0,      kt, lane);
    v16bf a1 = load_a_frag(A, NDIM, i0 + 16, kt, lane);
    v16bf b0 = load_b_frag(B, NDIM, jt +  0, kt, lane);
    v16bf b1 = load_b_frag(B, NDIM, jt + 16, kt, lane);
    v16bf b2 = load_b_frag(B, NDIM, jt + 32, kt, lane);
    v16bf b3 = load_b_frag(B, NDIM, jt + 48, kt, lane);
    acc[0][0] = wmma_bf16(a0, b0, acc[0][0]);
    acc[0][1] = wmma_bf16(a0, b1, acc[0][1]);
    acc[0][2] = wmma_bf16(a0, b2, acc[0][2]);
    acc[0][3] = wmma_bf16(a0, b3, acc[0][3]);
    acc[1][0] = wmma_bf16(a1, b0, acc[1][0]);
    acc[1][1] = wmma_bf16(a1, b1, acc[1][1]);
    acc[1][2] = wmma_bf16(a1, b2, acc[1][2]);
    acc[1][3] = wmma_bf16(a1, b3, acc[1][3]);
  }
  int half = lane >> 4, col = lane & 15;
  float* dstc = triT + (size_t)c * NNTOT;
#pragma unroll
  for (int mi = 0; mi < 2; ++mi) {
#pragma unroll
    for (int nt = 0; nt < 4; ++nt) {
      int j = jt + nt * 16 + col;
#pragma unroll
      for (int r = 0; r < 8; ++r) {
        int i = i0 + mi * 16 + r + half * 8;
        dstc[(size_t)i * NDIM + j] = acc[mi][nt][r];   // lanes -> consecutive j
      }
    }
  }
}

// ---------------------------------------------------------------- kernel 4
// Per 16-row tile: gather channel-major tri (thread = channel, 16 consecutive
// rows = contiguous 64B), transpose via LDS, LayerNorm, bf16, WMMA x p_out^T,
// multiply by gate, write f32 out [m][c].
__global__ void __launch_bounds__(128) k_out(const float* __restrict__ triT,
                                             const float* __restrict__ w,
                                             const float* __restrict__ b,
                                             const unsigned short* __restrict__ Wpout,
                                             const unsigned short* __restrict__ gate,
                                             float* __restrict__ out) {
  __shared__ float fl[16 * CDIM];
  __shared__ unsigned short xb[16 * CDIM];
  int t = threadIdx.x;
  size_t mbase = (size_t)blockIdx.x * 16;

  // Coalesced channel-major gather + LDS transpose.
  {
    const float4* src = (const float4*)(triT + (size_t)t * NNTOT + mbase);
#pragma unroll
    for (int q = 0; q < 4; ++q) {
      float4 vv = src[q];
      fl[(q * 4 + 0) * CDIM + t] = vv.x;
      fl[(q * 4 + 1) * CDIM + t] = vv.y;
      fl[(q * 4 + 2) * CDIM + t] = vv.z;
      fl[(q * 4 + 3) * CDIM + t] = vv.w;
    }
  }
  __syncthreads();

  int row = t >> 3;                    // 0..15 (4 rows per wave)
  int seg = (t & 7) * 16;              // 8 lanes x 16 channels = 128
  float v[16];
#pragma unroll
  for (int i = 0; i < 16; ++i) v[i] = fl[row * CDIM + seg + i];
  float s = 0.f, s2 = 0.f;
#pragma unroll
  for (int i = 0; i < 16; ++i) { s += v[i]; s2 += v[i] * v[i]; }
#pragma unroll
  for (int msk = 4; msk >= 1; msk >>= 1) {   // reduce across the 8 lanes of a row
    s  += __shfl_xor(s,  msk, 32);
    s2 += __shfl_xor(s2, msk, 32);
  }
  float mu   = s  * (1.0f / CDIM);
  float var  = s2 * (1.0f / CDIM) - mu * mu;
  float rstd = rsqrtf(var + 1e-5f);
#pragma unroll
  for (int i = 0; i < 16; ++i) {
    int c = seg + i;
    xb[row * CDIM + c] = f2bf_raw((v[i] - mu) * rstd * w[c] + b[c]);
  }
  __syncthreads();

  int wave = t >> 5, lane = t & 31;
  int n0 = wave * 32;                  // wave covers 32 output channels
  v8f a0 = {}, a1 = {};
#pragma unroll
  for (int kt = 0; kt < CDIM; kt += 32) {
    v16bf a = load_a_frag(xb, CDIM, 0, kt, lane);        // LDS (ds_load)
    a0 = wmma_bf16(a, load_b_frag(Wpout, CDIM, n0,      kt, lane), a0);
    a1 = wmma_bf16(a, load_b_frag(Wpout, CDIM, n0 + 16, kt, lane), a1);
  }
  int half = lane >> 4, col0 = lane & 15;
  v8f accs[2] = {a0, a1};
#pragma unroll
  for (int nt = 0; nt < 2; ++nt) {
    int cc = n0 + nt * 16 + col0;
#pragma unroll
    for (int r = 0; r < 8; ++r) {
      size_t mm = mbase + r + half * 8;
      float g = bf2f_raw(gate[mm * CDIM + cc]);
      out[mm * CDIM + cc] = accs[nt][r] * g;
    }
  }
}

// ---------------------------------------------------------------- launcher
extern "C" void kernel_launch(void* const* d_in, const int* in_sizes, int n_in,
                              void* d_out, int out_size, void* d_ws, size_t ws_size,
                              hipStream_t stream) {
  const float* x      = (const float*)d_in[0];
  const float* niw    = (const float*)d_in[1];
  const float* nib    = (const float*)d_in[2];
  const float* g_in_w = (const float*)d_in[3];
  const float* p_in_w = (const float*)d_in[4];
  const float* g_out_w= (const float*)d_in[5];
  const float* now    = (const float*)d_in[6];
  const float* nob    = (const float*)d_in[7];
  const float* p_out_w= (const float*)d_in[8];

  char* ws = (char*)d_ws;
  const size_t XN_OFF   = 0;                                  // bf16 NN*C
  const size_t AT_OFF   = XN_OFF   + (size_t)NNTOT * CDIM * 2;
  const size_t BT_OFF   = AT_OFF   + (size_t)NNTOT * CDIM * 2;
  const size_t GATE_OFF = BT_OFF   + (size_t)NNTOT * CDIM * 2;
  const size_t TRI_OFF  = GATE_OFF + (size_t)NNTOT * CDIM * 2;
  const size_t W_OFF    = TRI_OFF  + (size_t)NNTOT * CDIM * 4;

  unsigned short* xn   = (unsigned short*)(ws + XN_OFF);
  unsigned short* at   = (unsigned short*)(ws + AT_OFF);
  unsigned short* bt   = (unsigned short*)(ws + BT_OFF);
  unsigned short* gate = (unsigned short*)(ws + GATE_OFF);
  float*          triT = (float*)(ws + TRI_OFF);
  unsigned short* wbf  = (unsigned short*)(ws + W_OFF);

  k_prep <<<384, 256, 0, stream>>>(g_in_w, p_in_w, g_out_w, p_out_w, wbf);
  k_ln_in<<<NNTOT / 4, 128, 0, stream>>>(x, niw, nib, xn);
  k_proj <<<NNTOT / 64, 128, 0, stream>>>(xn, wbf, at, bt, gate);
  k_tri  <<<dim3(32, CDIM), 128, 0, stream>>>(at, bt, triT);
  k_out  <<<NNTOT / 16, 128, 0, stream>>>(triT, now, nob, wbf + 81920, gate,
                                          (float*)d_out);
}
